// CTGRU_53987738911250
// MI455X (gfx1250) — compile-verified, matrix operand order checked by reading, not run on hardware
//
#include <hip/hip_runtime.h>

// ---------------------------------------------------------------------------
// CT-GRU scan on MI455X (gfx1250): bf16 WMMA GEMMs with async-LDS B staging.
// ---------------------------------------------------------------------------

typedef __bf16 bf16_t;
typedef __attribute__((ext_vector_type(16))) __bf16 v16bf;
typedef __attribute__((ext_vector_type(8)))  float  v8f;
typedef __attribute__((ext_vector_type(4)))  unsigned int v4u;

constexpr int Nn = 1024, Hh = 512, Ss = 9, Ii = 128, Tt = 64, Kk = 512;
constexpr int XI = Ii + Hh;                 // 640
constexpr float HALF_LOG10 = 1.1512925464970229f;   // 0.5*ln(10)

__device__ __forceinline__ bf16_t f2bf(float f) {
  union { float f; unsigned u; } v; v.f = f;
  unsigned r = v.u + 0x7FFFu + ((v.u >> 16) & 1u);   // round-to-nearest-even
  unsigned short h = (unsigned short)(r >> 16);
  return __builtin_bit_cast(bf16_t, h);
}

// ---------------------------------------------------------------------------
// Pack a row-major f32 weight matrix W[Kdim][Ncols] into the CDNA5 16-bit
// B-fragment layout (ISA 7.12.2): per 16-col tile nt, per 32-K chunk kc,
// lane l holds column n = nt*16 + (l&15); halfs j=0..15 hold
// K = kc*32 + ((l>=16)?8:0) + (j<8 ? j : 8+j).  One contiguous 32B load/lane.
// ---------------------------------------------------------------------------
__global__ void pack_b_kernel(const float* __restrict__ W, bf16_t* __restrict__ out,
                              int Kdim, int Ncols) {
  int idx = blockIdx.x * blockDim.x + threadIdx.x;
  int total = Kdim * Ncols;
  if (idx >= total) return;
  int j     = idx & 15;
  int lane  = (idx >> 4) & 31;
  int chunk = idx >> 9;                  // chunk = nt*nkc + kc
  int nkc   = Kdim >> 5;
  int nt    = chunk / nkc;
  int kc    = chunk - nt * nkc;
  int kbase = (lane >= 16) ? 8 : 0;
  int kk    = kc * 32 + kbase + ((j < 8) ? j : (8 + j));
  int n     = nt * 16 + (lane & 15);
  out[idx]  = f2bf(W[(size_t)kk * Ncols + n]);
}

// ---------------------------------------------------------------------------
// Gather: xi = [X_obs | h[idx]] as bf16; also seed sig_in's X part and
// capture interval[k] = t - last_t[idx[k]] BEFORE any state update.
// ---------------------------------------------------------------------------
__global__ void gather_kernel(const float* __restrict__ X_t, const int* __restrict__ idx_t,
                              const float* __restrict__ h, const float* __restrict__ last_t,
                              const float* __restrict__ obs_times, int ti,
                              bf16_t* __restrict__ xi_bf, bf16_t* __restrict__ sig_bf,
                              float* __restrict__ interval) {
  int i = blockIdx.x * blockDim.x + threadIdx.x;   // [0, K*XI)
  if (i >= Kk * XI) return;
  int k = i / XI;
  int c = i - k * XI;
  float v;
  if (c < Ii) { v = X_t[(size_t)k * Ii + c]; sig_bf[i] = f2bf(v); }
  else        { v = h[(size_t)idx_t[k] * Hh + (c - Ii)]; }
  xi_bf[i] = f2bf(v);
  if (c == 0) interval[k] = obs_times[ti] - last_t[idx_t[k]];
}

// ---------------------------------------------------------------------------
// WMMA GEMM: C[M,N] = act(A[M,Kdim] @ Bpk + bias).
//  * 4 waves/block; block tile = 128 rows x 64 cols; wave tile = 32 x 64.
//  * B fragments double-buffered in LDS via global_load_async_to_lds_b128
//    (each wave stages its own 16-col tile; all waves read all 4 tiles).
//  * Padded LDS lane stride (48B) for conflict-free 32B ds reads.
// ACT: 0=none, 1=relu, 2=tanh.  OUT_BF selects bf16 vs f32 output.
// ---------------------------------------------------------------------------
template<int ACT, bool OUT_BF>
__global__ __launch_bounds__(128)
void gemm_wmma_kernel(const bf16_t* __restrict__ A, int lda,
                      const bf16_t* __restrict__ Bpk,
                      const float* __restrict__ bias,
                      float* __restrict__ Cf, bf16_t* __restrict__ Cbf, int ldc,
                      int Kdim) {
  constexpr int LSTR = 24;                       // halfs per lane slot (16 data + 8 pad)
  __shared__ __align__(32) bf16_t sB[2][4][32 * LSTR];

  const int lane  = threadIdx.x & 31;
  const int wave  = threadIdx.x >> 5;
  const int n0    = blockIdx.x * 64;             // 4 x 16-col tiles
  const int m0    = blockIdx.y * 128 + wave * 32;
  const int nkc   = Kdim >> 5;
  const int kbase = (lane >= 16) ? 8 : 0;
  const int mrow  = m0 + (lane & 15);

  // Global byte offset of this wave's staged tile (tile index = wave).
  const unsigned tile_off = (unsigned)(((n0 >> 4) + wave) * nkc) * 1024u
                          + (unsigned)lane * 32u;
  const unsigned lds_slot = (unsigned)(uintptr_t)(&sB[0][wave][lane * LSTR]);
  constexpr unsigned LDS_BUF_STRIDE = 4u * 32u * LSTR * 2u;   // bytes

  auto stage = [&](int kc, int buf) {
    unsigned voff = tile_off + (unsigned)kc * 1024u;          // global byte offset
    unsigned ldst = lds_slot + (unsigned)buf * LDS_BUF_STRIDE;
    asm volatile("global_load_async_to_lds_b128 %0, %1, %2 offset:0"
                 :: "v"(ldst), "v"(voff), "s"(Bpk) : "memory");
    asm volatile("global_load_async_to_lds_b128 %0, %1, %2 offset:16"
                 :: "v"(ldst), "v"(voff), "s"(Bpk) : "memory");
  };

  v8f acc[2][4] = {};
  const bf16_t* ap0 = A + (size_t)mrow * lda + kbase;   // rows m0 .. m0+15
  const bf16_t* ap1 = ap0 + (size_t)16 * lda;           // rows m0+16 .. m0+31

  stage(0, 0);
  for (int kc = 0; kc < nkc; ++kc) {
    const int buf = kc & 1;
    asm volatile("s_wait_asynccnt 0x0" ::: "memory");   // my staged chunk landed
    __syncthreads();                                    // everyone's chunk landed
    if (kc + 1 < nkc) stage(kc + 1, buf ^ 1);

    // A fragments (ISA 16-bit A layout): two contiguous 16B chunks per lane.
    union { v16bf v; v4u u[2]; } a0, a1;
    a0.u[0] = *(const v4u*)(ap0);  a0.u[1] = *(const v4u*)(ap0 + 16);
    a1.u[0] = *(const v4u*)(ap1);  a1.u[1] = *(const v4u*)(ap1 + 16);
    ap0 += 32;  ap1 += 32;

#pragma unroll
    for (int nt = 0; nt < 4; ++nt) {
      v16bf b = *(const v16bf*)(&sB[buf][nt][lane * LSTR]);
      acc[0][nt] = __builtin_amdgcn_wmma_f32_16x16x32_bf16(
          false, a0.v, false, b, (short)0, acc[0][nt], false, false);
      acc[1][nt] = __builtin_amdgcn_wmma_f32_16x16x32_bf16(
          false, a1.v, false, b, (short)0, acc[1][nt], false, false);
    }
  }

  const int rowoff = (lane >= 16) ? 8 : 0;
#pragma unroll
  for (int mi = 0; mi < 2; ++mi) {
#pragma unroll
    for (int nt = 0; nt < 4; ++nt) {
      int ncol = n0 + nt * 16 + (lane & 15);
      float bv = bias ? bias[ncol] : 0.0f;
#pragma unroll
      for (int r = 0; r < 8; ++r) {
        int row = m0 + mi * 16 + r + rowoff;
        float v = acc[mi][nt][r] + bv;
        if (ACT == 1) v = fmaxf(v, 0.0f);
        if (ACT == 2) v = tanhf(v);
        if (OUT_BF) Cbf[(size_t)row * ldc + ncol] = f2bf(v);
        else        Cf[(size_t)row * ldc + ncol]  = v;
      }
    }
  }
}

// ---------------------------------------------------------------------------
// Masked MAE partial sums (LDS tree reduction + global float atomics).
// ---------------------------------------------------------------------------
__global__ void loss_kernel(const float* __restrict__ X_t, const float* __restrict__ M_t,
                            const float* __restrict__ P, float* __restrict__ acc) {
  __shared__ float sl[256];
  __shared__ float sm[256];
  int i = blockIdx.x * 256 + threadIdx.x;           // K*I exactly
  float m = M_t[i];
  float l = fabsf(X_t[i] - P[i]) * m;
  sl[threadIdx.x] = l;
  sm[threadIdx.x] = m;
  __syncthreads();
  for (int s = 128; s > 0; s >>= 1) {
    if (threadIdx.x < s) {
      sl[threadIdx.x] += sl[threadIdx.x + s];
      sm[threadIdx.x] += sm[threadIdx.x + s];
    }
    __syncthreads();
  }
  if (threadIdx.x == 0) {
    atomicAdd(&acc[0], sl[0]);
    atomicAdd(&acc[1], sm[0]);
  }
}

// ---------------------------------------------------------------------------
// Retrieval mix: r = softmax_S(-(R - log_tau)^2); sig_in[:,I+h] = sum_s r*hh.
// ---------------------------------------------------------------------------
__global__ void mix_kernel(const float* __restrict__ R, const float* __restrict__ hhat,
                           const int* __restrict__ idx_t, bf16_t* __restrict__ sig_bf) {
  int i = blockIdx.x * blockDim.x + threadIdx.x;    // [0, K*H)
  if (i >= Kk * Hh) return;
  int k  = i >> 9;
  int hc = i & (Hh - 1);
  const float* rp = R + (size_t)k * (Hh * Ss) + (size_t)hc * Ss;
  float e[Ss]; float mx = -1e30f;
#pragma unroll
  for (int s = 0; s < Ss; ++s) {
    float d = rp[s] - (float)s * HALF_LOG10;
    e[s] = -d * d;
    mx = fmaxf(mx, e[s]);
  }
  float sum = 0.0f;
#pragma unroll
  for (int s = 0; s < Ss; ++s) { e[s] = __expf(e[s] - mx); sum += e[s]; }
  const float* hp = hhat + ((size_t)idx_t[k] * Hh + hc) * Ss;
  float a = 0.0f;
#pragma unroll
  for (int s = 0; s < Ss; ++s) a += e[s] * hp[s];
  sig_bf[(size_t)k * XI + Ii + hc] = f2bf(a / sum);
}

// ---------------------------------------------------------------------------
// State update: z = softmax_S(-(Slog - log_tau)^2); decay; scatter.
// h is scattered with the PRE-update hh sum (faithful to reference).
// ---------------------------------------------------------------------------
__global__ void update_kernel(const float* __restrict__ Slog, const float* __restrict__ Htil,
                              const float* __restrict__ interval, const int* __restrict__ idx_t,
                              const float* __restrict__ obs_times, int ti,
                              float* __restrict__ hhat, float* __restrict__ h,
                              float* __restrict__ last_t) {
  int i = blockIdx.x * blockDim.x + threadIdx.x;    // [0, K*H)
  if (i >= Kk * Hh) return;
  int k  = i >> 9;
  int hc = i & (Hh - 1);
  const float* sp = Slog + (size_t)k * (Hh * Ss) + (size_t)hc * Ss;
  float z[Ss]; float mx = -1e30f;
#pragma unroll
  for (int s = 0; s < Ss; ++s) {
    float d = sp[s] - (float)s * HALF_LOG10;
    z[s] = -d * d;
    mx = fmaxf(mx, z[s]);
  }
  float sum = 0.0f;
#pragma unroll
  for (int s = 0; s < Ss; ++s) { z[s] = __expf(z[s] - mx); sum += z[s]; }
  float inv = 1.0f / sum;
  float ht  = Htil[(size_t)k * Hh + hc];            // already tanh'ed by GEMM epilogue
  int n     = idx_t[k];
  float* hp = hhat + ((size_t)n * Hh + hc) * Ss;
  float dt  = interval[k];
  float hsum = 0.0f;
#pragma unroll
  for (int s = 0; s < Ss; ++s) {
    float zi  = z[s] * inv;
    float old = hp[s];
    hsum += old;                                    // pre-update sum
    float inv_tau = __expf(-(float)s * HALF_LOG10); // 10^(-s/2)
    float dec = __expf(-dt * inv_tau);
    hp[s] = ((1.0f - zi) * old + zi * ht) * dec;
  }
  h[(size_t)n * Hh + hc] = hsum;
  if (hc == 0) last_t[n] = obs_times[ti];
}

__global__ void finalize_kernel(const float* __restrict__ acc, float* __restrict__ out) {
  if (threadIdx.x == 0) {
    out[0] = acc[0];
    out[1] = acc[0] / acc[1];
  }
}

// ---------------------------------------------------------------------------
extern "C" void kernel_launch(void* const* d_in, const int* in_sizes, int n_in,
                              void* d_out, int out_size, void* d_ws, size_t ws_size,
                              hipStream_t stream) {
  (void)in_sizes; (void)n_in; (void)out_size; (void)ws_size;
  const float* obs_times = (const float*)d_in[0];
  const int*   batch_idx = (const int*)  d_in[1];
  const float* X     = (const float*)d_in[2];
  const float* M     = (const float*)d_in[3];
  const float* W_r   = (const float*)d_in[4];
  const float* b_r   = (const float*)d_in[5];
  const float* W_sig = (const float*)d_in[6];
  const float* b_sig = (const float*)d_in[7];
  const float* W_st  = (const float*)d_in[8];
  const float* b_st  = (const float*)d_in[9];
  const float* Wp1   = (const float*)d_in[10];
  const float* bp1   = (const float*)d_in[11];
  const float* Wp2   = (const float*)d_in[12];
  const float* bp2   = (const float*)d_in[13];

  char* ws = (char*)d_ws;
  size_t off = 0;
  auto carve = [&](size_t bytes) -> void* {
    void* p = ws + off;
    off = (off + bytes + 255) & ~(size_t)255;
    return p;
  };

  bf16_t* Wr_pk   = (bf16_t*)carve((size_t)XI * Hh * Ss * 2);
  bf16_t* Wst_pk  = (bf16_t*)carve((size_t)XI * Hh * Ss * 2);
  bf16_t* Wsig_pk = (bf16_t*)carve((size_t)XI * Hh * 2);
  bf16_t* Wp1_pk  = (bf16_t*)carve((size_t)Hh * Hh * 2);
  bf16_t* Wp2_pk  = (bf16_t*)carve((size_t)Hh * Ii * 2);
  float* h_state    = (float*)carve((size_t)Nn * Hh * 4);
  float* hhat_state = (float*)carve((size_t)Nn * Hh * Ss * 4);
  float* last_t     = (float*)carve((size_t)Nn * 4);
  float* acc        = (float*)carve(2 * sizeof(float));
  size_t zero_end   = off;
  bf16_t* xi_bf   = (bf16_t*)carve((size_t)Kk * XI * 2);
  bf16_t* sig_bf  = (bf16_t*)carve((size_t)Kk * XI * 2);
  bf16_t* phid_bf = (bf16_t*)carve((size_t)Kk * Hh * 2);
  float* p_out    = (float*)carve((size_t)Kk * Ii * 4);
  float* R_out    = (float*)carve((size_t)Kk * Hh * Ss * 4);
  float* S_out    = (float*)carve((size_t)Kk * Hh * Ss * 4);
  float* htilde   = (float*)carve((size_t)Kk * Hh * 4);
  float* interval = (float*)carve((size_t)Kk * 4);

  // Zero persistent state (h, h_hat, last_t, loss accumulators) every call.
  size_t zero_start = (size_t)((char*)h_state - ws);
  hipMemsetAsync(ws + zero_start, 0, zero_end - zero_start, stream);

  // One-time (per launch) weight conversion + fragment packing.
  auto pack = [&](const float* W, bf16_t* dst, int Kd, int Nc) {
    int total = Kd * Nc;
    pack_b_kernel<<<(total + 255) / 256, 256, 0, stream>>>(W, dst, Kd, Nc);
  };
  pack(W_r,   Wr_pk,   XI, Hh * Ss);
  pack(W_st,  Wst_pk,  XI, Hh * Ss);
  pack(W_sig, Wsig_pk, XI, Hh);
  pack(Wp1,   Wp1_pk,  Hh, Hh);
  pack(Wp2,   Wp2_pk,  Hh, Ii);

  for (int t = 0; t < Tt; ++t) {
    const float* X_t  = X + (size_t)t * Kk * Ii;
    const float* M_t  = M + (size_t)t * Kk * Ii;
    const int*   idxt = batch_idx + (size_t)t * Kk;

    gather_kernel<<<(Kk * XI + 255) / 256, 256, 0, stream>>>(
        X_t, idxt, h_state, last_t, obs_times, t, xi_bf, sig_bf, interval);

    // p_model hidden = relu(h_i @ Wp1 + bp1)  -> bf16
    gemm_wmma_kernel<1, true><<<dim3(Hh / 64, Kk / 128), 128, 0, stream>>>(
        xi_bf + Ii, XI, Wp1_pk, bp1, nullptr, phid_bf, Hh, Hh);
    // p = hidden @ Wp2 + bp2  -> f32
    gemm_wmma_kernel<0, false><<<dim3(Ii / 64, Kk / 128), 128, 0, stream>>>(
        phid_bf, Hh, Wp2_pk, bp2, p_out, nullptr, Ii, Hh);

    loss_kernel<<<(Kk * Ii) / 256, 256, 0, stream>>>(X_t, M_t, p_out, acc);

    // log_tau_R = xi @ W_r + b_r   (512 x 640 x 4608)
    gemm_wmma_kernel<0, false><<<dim3((Hh * Ss) / 64, Kk / 128), 128, 0, stream>>>(
        xi_bf, XI, Wr_pk, b_r, R_out, nullptr, Hh * Ss, XI);
    // log_tau_S = xi @ W_st + b_st
    gemm_wmma_kernel<0, false><<<dim3((Hh * Ss) / 64, Kk / 128), 128, 0, stream>>>(
        xi_bf, XI, Wst_pk, b_st, S_out, nullptr, Hh * Ss, XI);

    mix_kernel<<<(Kk * Hh) / 256, 256, 0, stream>>>(R_out, hhat_state, idxt, sig_bf);

    // h_tilde = tanh(sig_in @ W_sig + b_sig)
    gemm_wmma_kernel<2, false><<<dim3(Hh / 64, Kk / 128), 128, 0, stream>>>(
        sig_bf, XI, Wsig_pk, b_sig, htilde, nullptr, Hh, XI);

    update_kernel<<<(Kk * Hh) / 256, 256, 0, stream>>>(
        S_out, htilde, interval, idxt, obs_times, t, hhat_state, h_state, last_t);
  }

  finalize_kernel<<<1, 32, 0, stream>>>(acc, (float*)d_out);
}
